// ResultParser_43645457662371
// MI455X (gfx1250) — compile-verified, matrix operand order checked by reading, not run on hardware
//
#include <hip/hip_runtime.h>
#include <hip/hip_bf16.h>

typedef __attribute__((ext_vector_type(16))) _Float16 v16h;
typedef __attribute__((ext_vector_type(8)))  float    v8f;

#define NGT   128   // B*GT_PER_IMG
#define NPRED 512   // B*PRED_PER_IMG
#define MAPD  64
#define HW    4096  // 64*64
#define NCH   400

// ---------------------------------------------------------------------------
// Stage 1: GT<->pred matching via WMMA distance GEMM (128x512, K=2 padded to 32)
// 1 block, 256 threads = 8 waves; wave w owns GT rows [16w, 16w+16).
// d^2 = |g|^2 + |p|^2 - 2*(gy*py + gx*px); coords <= 63 are exact in f16 and
// wmma f16->f32 products are full precision, so distances are exact integers
// => argmin (with first-index tie-break) matches the reference bit-exactly.
// ---------------------------------------------------------------------------
__global__ __launch_bounds__(256) void match_kernel(
    const int*   __restrict__ vgt_batch_ids,
    const int*   __restrict__ vgt_centers,   // (128,2) [y,x]
    const int*   __restrict__ pred_batch_ids,
    const int*   __restrict__ pred_cyxs,     // (512,2) [y,x]
    const float* __restrict__ offset_maps,   // (32,2,64,64)
    float*       __restrict__ out,           // tail: centers_pred at 102400
    int*         __restrict__ ws_flat,       // [128]
    float*       __restrict__ ws_y,          // [128]
    float*       __restrict__ ws_x)          // [128]
{
    const int lane  = threadIdx.x & 31;
    const int wave  = threadIdx.x >> 5;   // 0..7
    const int m0    = wave * 16;
    const int lhalf = lane >> 4;          // 0 or 1
    const int l16   = lane & 15;

    // A fragment (16x32 f16, MxK): lanes 0-15 hold M=l16, K=0..7 in VGPR0..3
    // packed pairs -> a[0]=K0, a[1]=K1. Lanes 16-31 hold K=8..15 -> all zero.
    v16h a = {};
    if (lhalf == 0) {
        int m = m0 + l16;
        a[0] = (_Float16)(float)vgt_centers[2 * m + 0];
        a[1] = (_Float16)(float)vgt_centers[2 * m + 1];
    }

    // C/D layout: lane holds rows M = m0 + lhalf*8 + r for VGPR r=0..7.
    float g2[8]; int gb[8];
    #pragma unroll
    for (int r = 0; r < 8; ++r) {
        int m = m0 + lhalf * 8 + r;
        float gy = (float)vgt_centers[2 * m + 0];
        float gx = (float)vgt_centers[2 * m + 1];
        g2[r] = gy * gy + gx * gx;
        gb[r] = vgt_batch_ids[m];
    }

    float best[8]; int bestn[8];
    #pragma unroll
    for (int r = 0; r < 8; ++r) { best[r] = __builtin_inff(); bestn[r] = 0x7fffffff; }

    for (int tile = 0; tile < NPRED / 16; ++tile) {
        const int n = tile * 16 + l16;             // this lane's N column
        float py = (float)pred_cyxs[2 * n + 0];
        float px = (float)pred_cyxs[2 * n + 1];
        int   pb = pred_batch_ids[n];
        float p2 = py * py + px * px;

        // B fragment (32x16 f16, KxN): lanes 0-15 hold N=l16, K=0,1 in b[0],b[1].
        v16h b = {};
        if (lhalf == 0) { b[0] = (_Float16)py; b[1] = (_Float16)px; }

        v8f c = {};
        c = __builtin_amdgcn_wmma_f32_16x16x32_f16(
                /*neg_a=*/false, a, /*neg_b=*/false, b,
                /*c_mod=*/(short)0, c, /*reuse_a=*/false, /*reuse_b=*/false);

        #pragma unroll
        for (int r = 0; r < 8; ++r) {
            float d2 = g2[r] + p2 - 2.0f * c[r];
            float v  = (pb == gb[r]) ? d2 : __builtin_inff();
            if (v < best[r] || (v == best[r] && n < bestn[r])) {
                best[r] = v; bestn[r] = n;
            }
        }
    }

    // argmin across the 16 lanes sharing each row-half (xor keeps bit4 intact).
    #pragma unroll
    for (int off = 8; off >= 1; off >>= 1) {
        #pragma unroll
        for (int r = 0; r < 8; ++r) {
            float ov = __shfl_xor(best[r],  off, 32);
            int   on = __shfl_xor(bestn[r], off, 32);
            if (ov < best[r] || (ov == best[r] && on < bestn[r])) {
                best[r] = ov; bestn[r] = on;
            }
        }
    }

    if (l16 == 0) {
        #pragma unroll
        for (int r = 0; r < 8; ++r) {
            int m = m0 + lhalf * 8 + r;
            int j = bestn[r];
            int cy = pred_cyxs[2 * j + 0]; cy = min(max(cy, 0), MAPD - 1);
            int cx = pred_cyxs[2 * j + 1]; cx = min(max(cx, 0), MAPD - 1);
            int flat = cy * MAPD + cx;
            ws_flat[m] = flat;
            int bidx = vgt_batch_ids[m];
            float off0 = offset_maps[(bidx * 2 + 0) * HW + flat];
            float off1 = offset_maps[(bidx * 2 + 1) * HW + flat];
            ws_y[m] = (float)cy + off0;
            ws_x[m] = (float)cx + off1;
            // centers_pred = [flat % MAP, flat // MAP] = [cx, cy]
            out[NGT * (2 * NCH) + 2 * m + 0] = (float)cx;
            out[NGT * (2 * NCH) + 2 * m + 1] = (float)cy;
        }
    }
}

// ---------------------------------------------------------------------------
// Stage 2: latency-bound gather. One thread per (row, channel): 3x3 max +
// bilinear sample. 51200 threads x 13 independent loads keeps plenty of
// requests in flight on 23.3 TB/s HBM (total footprint ~MBs -> L2 resident).
// ---------------------------------------------------------------------------
__global__ __launch_bounds__(256) void gather_kernel(
    const float* __restrict__ params,        // (32,400,64,64)
    const int*   __restrict__ vgt_batch_ids,
    const int*   __restrict__ ws_flat,
    const float* __restrict__ ws_y,
    const float* __restrict__ ws_x,
    float*       __restrict__ out)           // (128, 800) + tail
{
    int t = blockIdx.x * blockDim.x + threadIdx.x;
    if (t >= NGT * NCH) return;
    int row = t / NCH;
    int c   = t - row * NCH;

    int b    = vgt_batch_ids[row];
    int flat = ws_flat[row];
    int cy = flat >> 6, cx = flat & (MAPD - 1);
    const float* bp = params + ((size_t)b * NCH + c) * HW;

    // 3x3 clipped neighborhood max
    float mx = -__builtin_inff();
    #pragma unroll
    for (int dy = -1; dy <= 1; ++dy) {
        int yy = min(max(cy + dy, 0), MAPD - 1);
        #pragma unroll
        for (int dx = -1; dx <= 1; ++dx) {
            int xx = min(max(cx + dx, 0), MAPD - 1);
            mx = fmaxf(mx, bp[yy * MAPD + xx]);
        }
    }

    // bilinear sample at offset-adjusted center
    float y = ws_y[row], x = ws_x[row];
    float y0 = floorf(y), x0 = floorf(x);
    float wy1 = y - y0, wx1 = x - x0;
    float wy0 = 1.0f - wy1, wx0 = 1.0f - wx1;

    auto corner = [&](float yy, float xx) -> float {
        bool inb = (yy >= 0.0f) && (yy <= (float)(MAPD - 1)) &&
                   (xx >= 0.0f) && (xx <= (float)(MAPD - 1));
        int yi = (int)fminf(fmaxf(yy, 0.0f), (float)(MAPD - 1));
        int xi = (int)fminf(fmaxf(xx, 0.0f), (float)(MAPD - 1));
        float v = bp[yi * MAPD + xi];
        return inb ? v : 0.0f;
    };

    float bil = wy0 * wx0 * corner(y0,        x0)
              + wy0 * wx1 * corner(y0,        x0 + 1.0f)
              + wy1 * wx0 * corner(y0 + 1.0f, x0)
              + wy1 * wx1 * corner(y0 + 1.0f, x0 + 1.0f);

    out[(size_t)row * (2 * NCH) + c]       = bil;
    out[(size_t)row * (2 * NCH) + NCH + c] = mx;
}

extern "C" void kernel_launch(void* const* d_in, const int* in_sizes, int n_in,
                              void* d_out, int out_size, void* d_ws, size_t ws_size,
                              hipStream_t stream) {
    const float* params   = (const float*)d_in[0];
    const float* offsets  = (const float*)d_in[1];
    const int*   vgt_bid  = (const int*)d_in[2];
    const int*   vgt_ctr  = (const int*)d_in[3];
    const int*   pred_bid = (const int*)d_in[4];
    const int*   pred_cyx = (const int*)d_in[5];
    float* out = (float*)d_out;

    int*   wflat = (int*)d_ws;
    float* wy    = (float*)d_ws + NGT;
    float* wx    = (float*)d_ws + 2 * NGT;

    match_kernel<<<1, 256, 0, stream>>>(vgt_bid, vgt_ctr, pred_bid, pred_cyx,
                                        offsets, out, wflat, wy, wx);

    const int total = NGT * NCH;                 // 51200
    gather_kernel<<<(total + 255) / 256, 256, 0, stream>>>(
        params, vgt_bid, wflat, wy, wx, out);
}
